// SpectralEncoder_19421842113207
// MI455X (gfx1250) — matrix operand order repaired and checked by reading, not compiled, not run on hardware
//
#include <hip/hip_runtime.h>
#include <hip/hip_bf16.h>
#include <cstdint>

#define N_NODES 100000
#define E_EDGES 1600000
#define IN_DIM  128
#define PE_DIM  16
#define HID     128
#define LAT     64
#define D0      144   // IN_DIM + PE_DIM
#define LD0     160   // padded to multiple of 32 (WMMA K)
#define LD1     128
#define WAVES   8     // waves per GEMM block

typedef __attribute__((ext_vector_type(16))) __bf16    v16bf;
typedef __attribute__((ext_vector_type(8)))  float     v8f;
typedef __attribute__((ext_vector_type(4)))  unsigned  u32x4;
typedef __attribute__((ext_vector_type(8)))  int       i32x8;
typedef __attribute__((ext_vector_type(4)))  int       i32x4;

union FragBF { unsigned u[8]; v16bf v; };

// f32 pair -> packed bf16, round-half-up (differs from RNE only on exact ties).
__device__ __forceinline__ unsigned pack2(float lo, float hi) {
  unsigned a = __float_as_uint(lo) + 0x8000u;
  unsigned b = __float_as_uint(hi) + 0x8000u;
  return (a >> 16) | (b & 0xffff0000u);
}
__device__ __forceinline__ void atomAdd(float* p, float v) {
  __hip_atomic_fetch_add(p, v, __ATOMIC_RELAXED, __HIP_MEMORY_SCOPE_AGENT);
}
__device__ __forceinline__ void sched_fence() {
#if __has_builtin(__builtin_amdgcn_sched_barrier)
  __builtin_amdgcn_sched_barrier(0);     // nothing may be moved across
#endif
}

// ---------------- elementwise helpers ----------------
__global__ void k_fill(float* p, int n, float v) {
  int i = blockIdx.x * blockDim.x + threadIdx.x;
  if (i < n) p[i] = v;
}
__global__ void k_negcopy(float* dst, const float* src, int n) {
  int i = blockIdx.x * blockDim.x + threadIdx.x;
  if (i < n) dst[i] = -src[i];
}

// ---------------- graph prep ----------------
__global__ void k_deg(const int* src, const int* dst, const float* w, float* deg) {
  int e = blockIdx.x * blockDim.x + threadIdx.x;
  if (e >= E_EDGES) return;
  int s = src[e], d = dst[e];
  if (s == d) return;                 // self-loops dropped
  atomAdd(&deg[s], w[e]);
}
__global__ void k_dis(const float* deg, float* dis) {
  int i = blockIdx.x * blockDim.x + threadIdx.x;
  if (i >= N_NODES) return;
  float d = deg[i];
  dis[i] = (d > 0.f) ? rsqrtf(d) : 0.f;
}
__global__ void k_what(const int* src, const int* dst, const float* w,
                       const float* dis, float* what) {
  int e = blockIdx.x * blockDim.x + threadIdx.x;
  if (e >= E_EDGES) return;
  int s = src[e], d = dst[e];
  float ww = (s == d) ? 0.f : w[e];
  what[e] = -dis[s] * ww * dis[d];
}

// ---------------- concat [x | lap_pe | 0-pad] into ld=160 buffer ----------------
__global__ void k_concat(const float* x, const float* pe, float* t0) {
  int gid = blockIdx.x * blockDim.x + threadIdx.x;   // N * 40 float4 chunks
  if (gid >= N_NODES * (LD0 / 4)) return;
  int c   = gid % (LD0 / 4);
  int i   = gid / (LD0 / 4);
  int col = 4 * c;
  float4 v;
  if (col < IN_DIM)      v = *(const float4*)(x  + (long)i * IN_DIM + col);
  else if (col < D0)     v = *(const float4*)(pe + (long)i * PE_DIM + (col - IN_DIM));
  else                   v = make_float4(0.f, 0.f, 0.f, 0.f);
  *(float4*)(t0 + (long)i * LD0 + col) = v;
}

// ---------------- sparse propagation: y[dst] += alpha * w_hat[e] * x[src] ----------------
__global__ void k_prop(const int* src, const int* dst, const float* what,
                       const float* x, float* y, int ld, int chunks, float alpha) {
  long gid = (long)blockIdx.x * blockDim.x + threadIdx.x;
  long tot = (long)E_EDGES * chunks;
  if (gid >= tot) return;
  int  c = (int)(gid % chunks);
  long e = gid / chunks;
  float wv = what[e] * alpha;
  if (wv == 0.f) return;
  int s = src[e], d = dst[e];
  float4 xv = *(const float4*)(x + (long)s * ld + 4 * c);
  float* yp = y + (long)d * ld + 4 * c;
  atomAdd(yp + 0, wv * xv.x);
  atomAdd(yp + 1, wv * xv.y);
  atomAdd(yp + 2, wv * xv.z);
  atomAdd(yp + 3, wv * xv.w);
}

// ---------------- weight swizzle: f32 [Ksrc x Nout] -> bf16 B-fragments ----------------
// B 32x16 bf16 layout: lane l holds n = l%16; VGPR j packs K = kt*32 + (l<16?0:8) + 2(j&3) + (j<4?0:16) + {0,1}
__global__ void k_swz(const float* W, unsigned* dst, int Ksrc, int Nout,
                      int ktiles, int ntiles) {
  int gid = blockIdx.x * blockDim.x + threadIdx.x;
  int tot = ktiles * ntiles * 32;
  if (gid >= tot) return;
  int lane = gid & 31;
  int t    = gid >> 5;
  int nt   = t % ntiles;
  int kt   = t / ntiles;
  int n     = nt * 16 + (lane & 15);
  int kbase = kt * 32 + ((lane < 16) ? 0 : 8);
  unsigned o[8];
#pragma unroll
  for (int j = 0; j < 8; ++j) {
    int K0 = kbase + 2 * (j & 3) + ((j & 4) ? 16 : 0);
    float lo = (K0     < Ksrc) ? W[(long)K0 * Nout + n]       : 0.f;
    float hi = (K0 + 1 < Ksrc) ? W[(long)(K0 + 1) * Nout + n] : 0.f;
    o[j] = pack2(lo, hi);
  }
  uint4* dp = (uint4*)(dst + (long)gid * 8);
  dp[0] = make_uint4(o[0], o[1], o[2], o[3]);
  dp[1] = make_uint4(o[4], o[5], o[6], o[7]);
}

// --------- stage swizzled B (nd dwords) from global into LDS via TDM ---------
__device__ __forceinline__ void tdm_stage_b(const unsigned* Bf, unsigned nd) {
#if __has_builtin(__builtin_amdgcn_tensor_load_to_lds)
  unsigned long long ga = (unsigned long long)Bf;
  // D# group0: count=1, lds_addr=0, global_addr, type=2 (bits 127:126)
  u32x4 g0 = { 1u,
               0u,
               (unsigned)(ga & 0xffffffffu),
               (unsigned)((ga >> 32) & 0x01ffffffu) | 0x80000000u };
  // D# group1: data_size=2 (4B); tensor_dim0=nd; tensor_dim1=1; tile_dim0=nd;
  //            tile_dim1=1; tensor_dim0_stride=nd  (1 row of nd dwords)
  i32x8 g1 = { (int)0x00020000u,
               (int)((nd & 0xffffu) << 16),
               (int)(((nd >> 16) & 0xffffu) | (1u << 16)),
               (int)((nd & 0xffffu) << 16),
               1,
               (int)nd,
               0, 0 };
  i32x4 g2 = { 0, 0, 0, 0 };
  i32x4 g3 = { 0, 0, 0, 0 };
#if __clang_major__ >= 23
  i32x8 g1b = { 0, 0, 0, 0, 0, 0, 0, 0 };
  __builtin_amdgcn_tensor_load_to_lds(g0, g1, g2, g3, g1b, 0);
#else
  __builtin_amdgcn_tensor_load_to_lds(g0, g1, g2, g3, 0);
#endif
  __builtin_amdgcn_s_wait_tensorcnt(0);
#endif
}

// ---------------- WMMA GEMM: Cout = [Cacc +] A(f32->bf16) @ Bfrag (+bias, relu) ----------------
// One wave computes 16 rows x (NT*16) cols.  B staged in LDS via TDM.  Per k-tile,
// all NT B-fragments are ds-loaded into distinct registers and pinned above a
// sched_barrier so the WMMA chain runs on counted (overlapping) s_wait_dscnt.
template<int NT, int KT>
__global__ __launch_bounds__(WAVES * 32, 1) void k_gemm(const float* __restrict__ A, int lda,
                                                        const unsigned* __restrict__ Bf,
                                                        const float* __restrict__ bias,
                                                        const float* __restrict__ Cacc,
                                                        float* __restrict__ Cout, int ldc,
                                                        int relu, int M) {
  extern __shared__ unsigned lb[];               // KT*NT*256 dwords
  const int tid = threadIdx.x;
#if __has_builtin(__builtin_amdgcn_tensor_load_to_lds)
  if (tid < 32) tdm_stage_b(Bf, (unsigned)(KT * NT * 256));
#else
  for (int i = tid; i < KT * NT * 256; i += WAVES * 32) lb[i] = Bf[i];
#endif
  __syncthreads();

  const int lane = tid & 31;
  const int wave = tid >> 5;
  const int row0 = (blockIdx.x * WAVES + wave) * 16;
  if (row0 >= M) return;                         // wave-uniform: EXEC all-ones
  const int m    = lane & 15;
  const int hi   = lane >> 4;                    // 0 | 1
  const int ncol = lane & 15;

  v8f c[NT];
#pragma unroll
  for (int nt = 0; nt < NT; ++nt) {
    if (Cacc) {
      const float* cp = Cacc + (long)row0 * ldc + nt * 16 + ncol;
#pragma unroll
      for (int r = 0; r < 8; ++r) c[nt][r] = cp[(long)(r + 8 * hi) * ldc];
    } else {
      c[nt] = (v8f)0.f;
    }
  }

  const int c0 = hi * 8;                // A runs: [c0,c0+8) and [16+c0,16+c0+8)
  const float* abase = A + (long)(row0 + m) * lda;
  float4 na0 = *(const float4*)(abase + c0);
  float4 na1 = *(const float4*)(abase + c0 + 4);
  float4 na2 = *(const float4*)(abase + 16 + c0);
  float4 na3 = *(const float4*)(abase + 16 + c0 + 4);
#pragma unroll
  for (int kt = 0; kt < KT; ++kt) {
    // 1) issue every B-fragment load of this k-tile into distinct registers
    uint4 bb0[NT], bb1[NT];
    const uint4* bb = (const uint4*)lb + ((long)kt * NT * 32 + lane) * 2;
#pragma unroll
    for (int nt = 0; nt < NT; ++nt) { bb0[nt] = bb[nt * 64]; bb1[nt] = bb[nt * 64 + 1]; }

    float4 f0 = na0, f1 = na1, f2 = na2, f3 = na3;
    // 2) prefetch next A tile
    if (kt + 1 < KT) {
      const float* ap = abase + (kt + 1) * 32;
      na0 = *(const float4*)(ap + c0);
      na1 = *(const float4*)(ap + c0 + 4);
      na2 = *(const float4*)(ap + 16 + c0);
      na3 = *(const float4*)(ap + 16 + c0 + 4);
    }
    // 3) pin all loads above this point: they may not sink below
    sched_fence();
    // 4) A bf16 pack (covers LDS latency)
    FragBF a;
    a.u[0] = pack2(f0.x, f0.y); a.u[1] = pack2(f0.z, f0.w);
    a.u[2] = pack2(f1.x, f1.y); a.u[3] = pack2(f1.z, f1.w);
    a.u[4] = pack2(f2.x, f2.y); a.u[5] = pack2(f2.z, f2.w);
    a.u[6] = pack2(f3.x, f3.y); a.u[7] = pack2(f3.z, f3.w);
    // 5) WMMA chain with overlapping waits
#pragma unroll
    for (int nt = 0; nt < NT; ++nt) {
      FragBF b;
      b.u[0] = bb0[nt].x; b.u[1] = bb0[nt].y; b.u[2] = bb0[nt].z; b.u[3] = bb0[nt].w;
      b.u[4] = bb1[nt].x; b.u[5] = bb1[nt].y; b.u[6] = bb1[nt].z; b.u[7] = bb1[nt].w;
      c[nt] = __builtin_amdgcn_wmma_f32_16x16x32_bf16(
          false, a.v, false, b.v, (short)0, c[nt], false, false);
    }
  }

#pragma unroll
  for (int nt = 0; nt < NT; ++nt) {
    float bv = bias ? bias[nt * 16 + ncol] : 0.f;
    float* op = Cout + (long)row0 * ldc + nt * 16 + ncol;
#pragma unroll
    for (int r = 0; r < 8; ++r) {
      float v = c[nt][r] + bv;
      if (relu) v = fmaxf(v, 0.f);
      op[(long)(r + 8 * hi) * ldc] = v;
    }
  }
}

// ---------------- launch ----------------
static inline int nblk(long n, int b) { return (int)((n + b - 1) / b); }

extern "C" void kernel_launch(void* const* d_in, const int* in_sizes, int n_in,
                              void* d_out, int out_size, void* d_ws, size_t ws_size,
                              hipStream_t stream) {
  const float* x   = (const float*)d_in[0];
  const int*   ei  = (const int*)d_in[1];
  const float* pe  = (const float*)d_in[2];
  const float* ew  = (const float*)d_in[3];
  const float* W0  = (const float*)d_in[4];
  const float* b0  = (const float*)d_in[5];
  const float* W1  = (const float*)d_in[6];
  const float* b1  = (const float*)d_in[7];
  const float* Wmu = (const float*)d_in[8];
  const float* bmu = (const float*)d_in[9];
  const float* Wlv = (const float*)d_in[10];
  const float* blv = (const float*)d_in[11];
  float* dout = (float*)d_out;

  const int* src = ei;
  const int* dst = ei + E_EDGES;

  float* ws   = (float*)d_ws;
  float* deg  = ws;
  float* dis  = deg + N_NODES;
  float* what = dis + N_NODES;
  float* T0   = what + E_EDGES;
  float* T1   = T0 + (size_t)N_NODES * LD0;
  float* T2   = T1 + (size_t)N_NODES * LD0;
  float* acc  = T2 + (size_t)N_NODES * LD0;
  unsigned* w0f  = (unsigned*)(acc + (size_t)N_NODES * LD1);
  unsigned* w1f  = w0f + 4 * (5 * 8 * 32 * 8);
  unsigned* wmuf = w1f + 4 * (4 * 8 * 32 * 8);
  unsigned* wlvf = wmuf + (4 * 4 * 32 * 8);

  const int B = 256;
  // --- weight swizzle (K padded with zeros to ktiles*32) ---
  for (int k = 0; k < 4; ++k)
    k_swz<<<nblk(5 * 8 * 32, B), B, 0, stream>>>(W0 + (long)k * D0 * HID,
                                                 w0f + (long)k * 5 * 8 * 32 * 8, D0, HID, 5, 8);
  for (int k = 0; k < 4; ++k)
    k_swz<<<nblk(4 * 8 * 32, B), B, 0, stream>>>(W1 + (long)k * HID * HID,
                                                 w1f + (long)k * 4 * 8 * 32 * 8, HID, HID, 4, 8);
  k_swz<<<nblk(4 * 4 * 32, B), B, 0, stream>>>(Wmu, wmuf, HID, LAT, 4, 4);
  k_swz<<<nblk(4 * 4 * 32, B), B, 0, stream>>>(Wlv, wlvf, HID, LAT, 4, 4);

  // --- graph prep ---
  k_fill<<<nblk(N_NODES, B), B, 0, stream>>>(deg, N_NODES, 0.f);
  k_deg<<<nblk(E_EDGES, B), B, 0, stream>>>(src, dst, ew, deg);
  k_dis<<<nblk(N_NODES, B), B, 0, stream>>>(deg, dis);
  k_what<<<nblk(E_EDGES, B), B, 0, stream>>>(src, dst, ew, dis, what);

  const int rowsPerBlk = WAVES * 16;                 // 128
  const int gblk  = (N_NODES + rowsPerBlk - 1) / rowsPerBlk;
  const int lds85 = 5 * 8 * 1024;                    // KT=5, NT=8
  const int lds84 = 4 * 8 * 1024;                    // KT=4, NT=8
  const int lds44 = 4 * 4 * 1024;                    // KT=4, NT=4

  // ================= layer 0 (K=144 padded to 160) =================
  k_concat<<<nblk((long)N_NODES * (LD0 / 4), B), B, 0, stream>>>(x, pe, T0);
  k_gemm<8, 5><<<gblk, WAVES * 32, lds85, stream>>>(T0, LD0, w0f + 0 * 5 * 8 * 32 * 8,
                                                    nullptr, nullptr, acc, LD1, 0, N_NODES);
  // Tx1 = P(Tx0)
  k_fill<<<nblk((long)N_NODES * LD0, B), B, 0, stream>>>(T1, N_NODES * LD0, 0.f);
  k_prop<<<nblk((long)E_EDGES * (D0 / 4), B), B, 0, stream>>>(src, dst, what, T0, T1, LD0, D0 / 4, 1.f);
  k_gemm<8, 5><<<gblk, WAVES * 32, lds85, stream>>>(T1, LD0, w0f + 1 * 5 * 8 * 32 * 8,
                                                    nullptr, acc, acc, LD1, 0, N_NODES);
  // Tx2 = 2 P(Tx1) - Tx0
  k_negcopy<<<nblk((long)N_NODES * LD0, B), B, 0, stream>>>(T2, T0, N_NODES * LD0);
  k_prop<<<nblk((long)E_EDGES * (D0 / 4), B), B, 0, stream>>>(src, dst, what, T1, T2, LD0, D0 / 4, 2.f);
  k_gemm<8, 5><<<gblk, WAVES * 32, lds85, stream>>>(T2, LD0, w0f + 2 * 5 * 8 * 32 * 8,
                                                    nullptr, acc, acc, LD1, 0, N_NODES);
  // Tx3 = 2 P(Tx2) - Tx1 ; fused: h1 = relu(acc + Tx3@W0[3] + b0) -> T1 (ld 128)
  k_negcopy<<<nblk((long)N_NODES * LD0, B), B, 0, stream>>>(T0, T1, N_NODES * LD0);
  k_prop<<<nblk((long)E_EDGES * (D0 / 4), B), B, 0, stream>>>(src, dst, what, T2, T0, LD0, D0 / 4, 2.f);
  k_gemm<8, 5><<<gblk, WAVES * 32, lds85, stream>>>(T0, LD0, w0f + 3 * 5 * 8 * 32 * 8,
                                                    b0, acc, T1, LD1, 1, N_NODES);

  // ================= layer 1 (K=128) : Tx0 = T1 =================
  k_gemm<8, 4><<<gblk, WAVES * 32, lds84, stream>>>(T1, LD1, w1f + 0 * 4 * 8 * 32 * 8,
                                                    nullptr, nullptr, acc, LD1, 0, N_NODES);
  k_fill<<<nblk((long)N_NODES * LD1, B), B, 0, stream>>>(T2, N_NODES * LD1, 0.f);
  k_prop<<<nblk((long)E_EDGES * (LD1 / 4), B), B, 0, stream>>>(src, dst, what, T1, T2, LD1, LD1 / 4, 1.f);
  k_gemm<8, 4><<<gblk, WAVES * 32, lds84, stream>>>(T2, LD1, w1f + 1 * 4 * 8 * 32 * 8,
                                                    nullptr, acc, acc, LD1, 0, N_NODES);
  k_negcopy<<<nblk((long)N_NODES * LD1, B), B, 0, stream>>>(T0, T1, N_NODES * LD1);
  k_prop<<<nblk((long)E_EDGES * (LD1 / 4), B), B, 0, stream>>>(src, dst, what, T2, T0, LD1, LD1 / 4, 2.f);
  k_gemm<8, 4><<<gblk, WAVES * 32, lds84, stream>>>(T0, LD1, w1f + 2 * 4 * 8 * 32 * 8,
                                                    nullptr, acc, acc, LD1, 0, N_NODES);
  // Tx3 = 2 P(Tx2=T0) - Tx1=T2 ; fused: h2 = relu(acc + Tx3@W1[3] + b1) -> T2
  k_negcopy<<<nblk((long)N_NODES * LD1, B), B, 0, stream>>>(T1, T2, N_NODES * LD1);
  k_prop<<<nblk((long)E_EDGES * (LD1 / 4), B), B, 0, stream>>>(src, dst, what, T0, T1, LD1, LD1 / 4, 2.f);
  k_gemm<8, 4><<<gblk, WAVES * 32, lds84, stream>>>(T1, LD1, w1f + 3 * 4 * 8 * 32 * 8,
                                                    b1, acc, T2, LD1, 1, N_NODES);

  // ================= heads =================
  k_gemm<4, 4><<<gblk, WAVES * 32, lds44, stream>>>(T2, LD1, wmuf, bmu, nullptr,
                                                    dout, LAT, 0, N_NODES);
  k_gemm<4, 4><<<gblk, WAVES * 32, lds44, stream>>>(T2, LD1, wlvf, blv, nullptr,
                                                    dout + (size_t)N_NODES * LAT, LAT, 0, N_NODES);
}
